// ui_aggregator_24833500905766
// MI455X (gfx1250) — matrix operand order repaired and checked by reading, not compiled, not run on hardware
//
#include <hip/hip_runtime.h>
#include <hip/hip_bf16.h>

typedef __attribute__((ext_vector_type(16))) _Float16 v16h;
typedef __attribute__((ext_vector_type(8)))  float    v8f;

#define B_TOTAL 16384
#define H_NB    50
#define SXS     136   // 128-col LDS row stride (halfs): 272B, 16B aligned, conflict-friendly
#define SWS     72    // 64-col LDS row stride (halfs): 144B, 16B aligned

// f16 weight workspace layout (halfs), all stored transposed as [K][N]
#define OFF_WR1T 0
#define OFF_WA1T 8192
#define OFF_WR2T 16384
#define OFF_WA2T 20480
#define OFF_L1T  24576
#define OFF_A3   32768
#define WS_HALFS 32832

// ---------------- prep: fp32 weights -> f16, transposed [K][N] ----------------
__global__ void prep_weights(const float* __restrict__ w_r1_w, const float* __restrict__ att1_w,
                             const float* __restrict__ w_r2_w, const float* __restrict__ att2_w,
                             const float* __restrict__ lin1_w, const float* __restrict__ att3_w,
                             _Float16* __restrict__ ws) {
  int i = blockIdx.x * blockDim.x + threadIdx.x;
  if (i >= WS_HALFS) return;
  float v;
  if (i < 8192)       { int j = i;          v = w_r1_w[(j & 63) * 128 + (j >> 6)]; }
  else if (i < 16384) { int j = i - 8192;   v = att1_w[(j & 63) * 128 + (j >> 6)]; }
  else if (i < 20480) { int j = i - 16384;  v = w_r2_w[(j & 63) * 64  + (j >> 6)]; }
  else if (i < 24576) { int j = i - 20480;  v = att2_w[(j & 63) * 64  + (j >> 6)]; }
  else if (i < 32768) { int j = i - 24576;  v = lin1_w[(j & 63) * 128 + (j >> 6)]; }
  else                {                     v = att3_w[i - 32768]; }
  ws[i] = (_Float16)v;
}

// ---------------- WMMA fragment helpers (CDNA5 16-bit layouts) ----------------
// A (16x32 f16): lane l holds row (l&15); halves 0-7 = K[k0..k0+7], 8-15 = K[k0+16..k0+23],
// k0 = 8*(l>>4) within the 32-wide chunk  ->  two 16B LDS reads.
__device__ __forceinline__ v16h lds_afrag(const _Float16* p) {
  v16h a;
  uint4* d = reinterpret_cast<uint4*>(&a);
  d[0] = *reinterpret_cast<const uint4*>(p);
  d[1] = *reinterpret_cast<const uint4*>(p + 16);
  return a;
}
// B (32x16 f16): lane l holds column n = (l&15); halves 0-7 = K[k0..k0+7], 8-15 = K[k0+16..k0+23],
// k0 = 8*(l>>4) -> weights stored [K][N] row-major in LDS: strided scalar reads.
__device__ __forceinline__ v16h lds_bfrag(const _Float16* Wt, int kc, int nt, int lane) {
  const _Float16* p = Wt + (kc * 32 + ((lane >> 4) << 3)) * SWS + nt * 16 + (lane & 15);
  v16h b;
#pragma unroll
  for (int s = 0; s < 8; ++s) { b[s] = p[s * SWS]; b[8 + s] = p[(16 + s) * SWS]; }
  return b;
}
__device__ __forceinline__ v8f wmma_f16(v16h a, v16h b, v8f c) {
  return __builtin_amdgcn_wmma_f32_16x16x32_f16(false, a, false, b, (short)0, c, false, false);
}
// D (16x16 f32): lane l -> col nt*16+(l&15); VGPR v -> row mbase + v + 8*(l>>4)
__device__ __forceinline__ void store_relu(v8f c, _Float16* dst, int stride, int mbase, int nt, int lane) {
  int col = nt * 16 + (lane & 15);
  int rbase = mbase + ((lane >> 4) << 3);
  _Float16* p = dst + rbase * stride + col;
#pragma unroll
  for (int v = 0; v < 8; ++v) p[v * stride] = (_Float16)fmaxf(c[v], 0.0f);
}

// ---------------- fused aggregator ----------------
struct __align__(16) Smem {
  _Float16 wr1t[128 * SWS];
  _Float16 wa1t[128 * SWS];
  _Float16 wr2t[64 * SWS];
  _Float16 wa2t[64 * SWS];
  _Float16 l1t[128 * SWS];
  _Float16 a3[64];
  _Float16 u16[64];
  _Float16 x[64 * SXS];   // gathered [e_ui|e_r] f16; later reused for a2 activation
  _Float16 h[64 * SWS];   // h; later reused for att1 activation
  _Float16 o[64 * SWS];   // per-edge MLP output (persists for weighted sum)
  float bias[5 * 64];
  float a3b;
  float uf[64];
  float logit[64];        // logits; later reused for neigh vector
  float expv[64];
  int   idxI[64];
  int   idxR[64];
};

__global__ __launch_bounds__(128) void graphrec_agg(
    const int* __restrict__ nodes, const int* __restrict__ hui, const int* __restrict__ hr,
    const float* __restrict__ u2e, const float* __restrict__ i2e, const float* __restrict__ r2e,
    const float* __restrict__ b_r1, const float* __restrict__ b_r2,
    const float* __restrict__ b_a1, const float* __restrict__ b_a2,
    const float* __restrict__ b_a3, const float* __restrict__ b_l1,
    const _Float16* __restrict__ ws, float* __restrict__ out)
{
  __shared__ Smem sm;
  const int tid   = threadIdx.x;
  const int lane  = tid & 31;
  const int wv    = tid >> 5;      // wave 0..3 -> M tile
  const int mbase = wv * 16;
  const int mrow  = lane & 15;
  const int khalf = (lane >> 4) * 8;

  // one-time: f16 weights (d_ws, [K][64]) -> padded LDS ([K][SWS]); 512 rows x 8 uint4
  for (int i = tid; i < 512 * 8; i += 128) {
    int r = i >> 3, c = (i & 7) << 3;
    _Float16* ldst; const _Float16* gsrc;
    if (r < 128)      {               ldst = sm.wr1t + r  * SWS; gsrc = ws + OFF_WR1T + r  * 64; }
    else if (r < 256) { int rr=r-128; ldst = sm.wa1t + rr * SWS; gsrc = ws + OFF_WA1T + rr * 64; }
    else if (r < 320) { int rr=r-256; ldst = sm.wr2t + rr * SWS; gsrc = ws + OFF_WR2T + rr * 64; }
    else if (r < 384) { int rr=r-320; ldst = sm.wa2t + rr * SWS; gsrc = ws + OFF_WA2T + rr * 64; }
    else              { int rr=r-384; ldst = sm.l1t  + rr * SWS; gsrc = ws + OFF_L1T  + rr * 64; }
    *reinterpret_cast<uint4*>(ldst + c) = *reinterpret_cast<const uint4*>(gsrc + c);
  }
  if (tid < 64) {
    sm.a3[tid]         = ws[OFF_A3 + tid];
    sm.bias[tid]       = b_r1[tid];
    sm.bias[64 + tid]  = b_r2[tid];
    sm.bias[128 + tid] = b_a1[tid];
    sm.bias[192 + tid] = b_a2[tid];
    sm.bias[256 + tid] = b_l1[tid];
  }
  if (tid == 0) sm.a3b = b_a3[0];

  for (int b = blockIdx.x; b < B_TOTAL; b += gridDim.x) {
    __syncthreads();   // weights ready / previous iteration fully consumed

    if (tid < H_NB) { sm.idxI[tid] = hui[b * H_NB + tid]; sm.idxR[tid] = hr[b * H_NB + tid]; }
    if (tid < 64) {
      float uv = u2e[(size_t)nodes[b] * 64 + tid];
      sm.uf[tid]  = uv;
      sm.u16[tid] = (_Float16)uv;
    }
    __syncthreads();

    // gather x = [i2e[hui] | r2e[hr]] -> f16 LDS, rows >= 50 zero-padded
    for (int q = tid; q < 64 * 32; q += 128) {
      int row = q >> 5, part = q & 31, col = part << 2;
      float4 v = make_float4(0.f, 0.f, 0.f, 0.f);
      if (row < H_NB) {
        const float* src = (part < 16)
            ? (i2e + (size_t)sm.idxI[row] * 64 + col)
            : (r2e + (size_t)sm.idxR[row] * 64 + (col - 64));
        v = *reinterpret_cast<const float4*>(src);
      }
      union { _Float16 hh[4]; uint2 u; } t;
      t.hh[0] = (_Float16)v.x; t.hh[1] = (_Float16)v.y;
      t.hh[2] = (_Float16)v.z; t.hh[3] = (_Float16)v.w;
      *reinterpret_cast<uint2*>(sm.x + row * SXS + col) = t.u;
    }
    __syncthreads();

    // GEMM1: h = relu(x @ W1^T + b1), K=128
    {
      const _Float16* xr = sm.x + (mbase + mrow) * SXS + khalf;
      v16h a0 = lds_afrag(xr), a1 = lds_afrag(xr + 32), a2 = lds_afrag(xr + 64), a3f = lds_afrag(xr + 96);
#pragma unroll
      for (int nt = 0; nt < 4; ++nt) {
        float bv = sm.bias[nt * 16 + mrow];
        v8f c = {bv, bv, bv, bv, bv, bv, bv, bv};
        c = wmma_f16(a0,  lds_bfrag(sm.wr1t, 0, nt, lane), c);
        c = wmma_f16(a1,  lds_bfrag(sm.wr1t, 1, nt, lane), c);
        c = wmma_f16(a2,  lds_bfrag(sm.wr1t, 2, nt, lane), c);
        c = wmma_f16(a3f, lds_bfrag(sm.wr1t, 3, nt, lane), c);
        store_relu(c, sm.h, SWS, mbase, nt, lane);
      }
    }
    // GEMM2: o = relu(h @ W2^T + b2), K=64 (same wave's rows -> no barrier)
    {
      const _Float16* hrow = sm.h + (mbase + mrow) * SWS + khalf;
      v16h a0 = lds_afrag(hrow), a1 = lds_afrag(hrow + 32);
#pragma unroll
      for (int nt = 0; nt < 4; ++nt) {
        float bv = sm.bias[64 + nt * 16 + mrow];
        v8f c = {bv, bv, bv, bv, bv, bv, bv, bv};
        c = wmma_f16(a0, lds_bfrag(sm.wr2t, 0, nt, lane), c);
        c = wmma_f16(a1, lds_bfrag(sm.wr2t, 1, nt, lane), c);
        store_relu(c, sm.o, SWS, mbase, nt, lane);
      }
    }
    // ATT1: a = relu([o | u] @ A1^T + b), K=128; u chunks broadcast from LDS vector
    {
      const _Float16* orow = sm.o + (mbase + mrow) * SWS + khalf;
      v16h a0 = lds_afrag(orow), a1 = lds_afrag(orow + 32);
      v16h a2 = lds_afrag(sm.u16 + khalf), a3f = lds_afrag(sm.u16 + 32 + khalf);
#pragma unroll
      for (int nt = 0; nt < 4; ++nt) {
        float bv = sm.bias[128 + nt * 16 + mrow];
        v8f c = {bv, bv, bv, bv, bv, bv, bv, bv};
        c = wmma_f16(a0,  lds_bfrag(sm.wa1t, 0, nt, lane), c);
        c = wmma_f16(a1,  lds_bfrag(sm.wa1t, 1, nt, lane), c);
        c = wmma_f16(a2,  lds_bfrag(sm.wa1t, 2, nt, lane), c);
        c = wmma_f16(a3f, lds_bfrag(sm.wa1t, 3, nt, lane), c);
        store_relu(c, sm.h, SWS, mbase, nt, lane);   // a1 activation overwrites h
      }
    }
    // ATT2: a2 = relu(a1 @ A2^T + b), K=64 -> stored into sm.x (x no longer needed)
    {
      const _Float16* ar = sm.h + (mbase + mrow) * SWS + khalf;
      v16h a0 = lds_afrag(ar), a1 = lds_afrag(ar + 32);
#pragma unroll
      for (int nt = 0; nt < 4; ++nt) {
        float bv = sm.bias[192 + nt * 16 + mrow];
        v8f c = {bv, bv, bv, bv, bv, bv, bv, bv};
        c = wmma_f16(a0, lds_bfrag(sm.wa2t, 0, nt, lane), c);
        c = wmma_f16(a1, lds_bfrag(sm.wa2t, 1, nt, lane), c);
        store_relu(c, sm.x, SXS, mbase, nt, lane);
      }
    }
    __syncthreads();

    // logits = a2 @ att3^T + b3 (rows >= H_NB masked)
    if (tid < 64) {
      float acc = sm.a3b;
      const _Float16* ar = sm.x + tid * SXS;
#pragma unroll 8
      for (int k = 0; k < 64; ++k) acc += (float)ar[k] * (float)sm.a3[k];
      sm.logit[tid] = (tid < H_NB) ? acc : -1e30f;
    }
    __syncthreads();

    float mx = -1e30f;
    for (int r = 0; r < H_NB; ++r) mx = fmaxf(mx, sm.logit[r]);
    if (tid < H_NB) sm.expv[tid] = __expf(sm.logit[tid] - mx);
    __syncthreads();

    float s = 0.f;
    for (int r = 0; r < H_NB; ++r) s += sm.expv[r];
    float inv = 1.0f / s;
    if (tid < 64) {   // neigh[d] = sum_r att[r] * o[r][d]
      float acc = 0.f;
      for (int r = 0; r < H_NB; ++r) acc += sm.expv[r] * (float)sm.o[r * SWS + tid];
      sm.logit[tid] = acc * inv;     // reuse as neigh vector
    }
    __syncthreads();

    // out = relu([u | neigh] @ L1^T + b), f32, L1 transposed in LDS
    if (tid < 64) {
      float acc = sm.bias[256 + tid];
#pragma unroll 8
      for (int k = 0; k < 64; ++k) {
        acc += sm.uf[k]    * (float)sm.l1t[k * SWS + tid];
        acc += sm.logit[k] * (float)sm.l1t[(64 + k) * SWS + tid];
      }
      out[(size_t)b * 64 + tid] = fmaxf(acc, 0.f);
    }
  }
}

extern "C" void kernel_launch(void* const* d_in, const int* in_sizes, int n_in,
                              void* d_out, int out_size, void* d_ws, size_t ws_size,
                              hipStream_t stream) {
  const int*   nodes  = (const int*)d_in[0];
  const int*   hui    = (const int*)d_in[1];
  const int*   hr     = (const int*)d_in[2];
  const float* u2e    = (const float*)d_in[3];
  const float* i2e    = (const float*)d_in[4];
  const float* r2e    = (const float*)d_in[5];
  const float* w_r1_w = (const float*)d_in[6];
  const float* w_r1_b = (const float*)d_in[7];
  const float* w_r2_w = (const float*)d_in[8];
  const float* w_r2_b = (const float*)d_in[9];
  const float* att1_w = (const float*)d_in[10];
  const float* att1_b = (const float*)d_in[11];
  const float* att2_w = (const float*)d_in[12];
  const float* att2_b = (const float*)d_in[13];
  const float* att3_w = (const float*)d_in[14];
  const float* att3_b = (const float*)d_in[15];
  const float* lin1_w = (const float*)d_in[16];
  const float* lin1_b = (const float*)d_in[17];
  _Float16* ws  = (_Float16*)d_ws;
  float*    out = (float*)d_out;

  prep_weights<<<(WS_HALFS + 255) / 256, 256, 0, stream>>>(
      w_r1_w, att1_w, w_r2_w, att2_w, lin1_w, att3_w, ws);

  graphrec_agg<<<2048, 128, 0, stream>>>(
      nodes, hui, hr, u2e, i2e, r2e,
      w_r1_b, w_r2_b, att1_b, att2_b, att3_b, lin1_b,
      ws, out);
}
